// DSE_5428838662636
// MI455X (gfx1250) — matrix operand order repaired and checked by reading, not compiled, not run on hardware
//
#include <hip/hip_runtime.h>

typedef __attribute__((ext_vector_type(16))) _Float16 v16h;
typedef __attribute__((ext_vector_type(8)))  float    v8f;

#define Bsz 32
#define Hsz 128
#define Wsz 128
#define Csz 10

static constexpr size_t NPIX   = (size_t)Bsz * Hsz * Wsz;        // 524288
static constexpr size_t OFF_Y1  = 0;                              // f16 NPIX*32
static constexpr size_t OFF_PHI = OFF_Y1  + NPIX * 32 * 2;        // f16 NPIX*32
static constexpr size_t OFF_WB  = OFF_PHI + NPIX * 32 * 2;        // f32 NPIX*4
static constexpr size_t OFF_A0  = OFF_WB  + NPIX * 4 * 4;         // f32 NPIX*32
static constexpr size_t OFF_A1  = OFF_A0  + NPIX * 32 * 4;        // f32 NPIX*32
static constexpr size_t OFF_W2H = OFF_A1  + NPIX * 32 * 4;        // f16 9216
static constexpr size_t OFF_W3H = OFF_W2H + 9216 * 2;             // f16 1024
static constexpr size_t OFF_P   = OFF_W3H + 1024 * 2;             // f32 32*32*15

// ---------------- weight fp32 -> fp16 conversion ----------------
__global__ void k_prep(const float* __restrict__ w2, const float* __restrict__ w3,
                       _Float16* __restrict__ w2h, _Float16* __restrict__ w3h) {
    int t = blockIdx.x * blockDim.x + threadIdx.x;
    if (t < 9216) w2h[t] = (_Float16)w2[t];
    else if (t < 9216 + 1024) w3h[t - 9216] = (_Float16)w3[t - 9216];
}

// ---------------- conv1 (one-hot gather conv) + ReLU -> Y1 f16 ----------------
__global__ void k_conv1(const int* __restrict__ X, const float* __restrict__ w1,
                        const float* __restrict__ b1, _Float16* __restrict__ Y1) {
    __shared__ float sw1[9 * Csz * 32];   // 2880 floats
    __shared__ float sb1[32];
    for (int i = threadIdx.x; i < 9 * Csz * 32; i += blockDim.x) sw1[i] = w1[i];
    if (threadIdx.x < 32) sb1[threadIdx.x] = b1[threadIdx.x];
    __syncthreads();

    int p = blockIdx.x * blockDim.x + threadIdx.x;          // 0..NPIX-1
    int w = p & (Wsz - 1);
    int h = (p >> 7) & (Hsz - 1);
    int b = p >> 14;

    float4 acc[8];
    {
        const float4* bb = reinterpret_cast<const float4*>(sb1);
        #pragma unroll
        for (int j = 0; j < 8; ++j) acc[j] = bb[j];
    }
    #pragma unroll
    for (int ky = -1; ky <= 1; ++ky) {
        #pragma unroll
        for (int kx = -1; kx <= 1; ++kx) {
            int hh = h + ky, ww = w + kx;
            if ((unsigned)hh < (unsigned)Hsz && (unsigned)ww < (unsigned)Wsz) {
                int lab = X[((size_t)b * Hsz + hh) * Wsz + ww];
                const float4* row = reinterpret_cast<const float4*>(
                    &sw1[((ky + 1) * 3 + (kx + 1)) * Csz * 32 + lab * 32]);
                #pragma unroll
                for (int j = 0; j < 8; ++j) {
                    float4 r = row[j];
                    acc[j].x += r.x; acc[j].y += r.y; acc[j].z += r.z; acc[j].w += r.w;
                }
            }
        }
    }
    alignas(16) _Float16 hv[32];
    #pragma unroll
    for (int j = 0; j < 8; ++j) {
        float vx = acc[j].x > 0.f ? acc[j].x : 0.f;
        float vy = acc[j].y > 0.f ? acc[j].y : 0.f;
        float vz = acc[j].z > 0.f ? acc[j].z : 0.f;
        float vw = acc[j].w > 0.f ? acc[j].w : 0.f;
        hv[4 * j + 0] = (_Float16)vx; hv[4 * j + 1] = (_Float16)vy;
        hv[4 * j + 2] = (_Float16)vz; hv[4 * j + 3] = (_Float16)vw;
    }
    uint4* dst = reinterpret_cast<uint4*>(Y1 + (size_t)p * 32);
    const uint4* src = reinterpret_cast<const uint4*>(hv);
    dst[0] = src[0]; dst[1] = src[1]; dst[2] = src[2]; dst[3] = src[3];
}

// ---------------- conv2 (3x3 WMMA) + ReLU + conv3 (1x1 WMMA) -> phi f16 ----------------
union U16 { v16h v; uint4 u[2]; };

__global__ void k_conv23(const _Float16* __restrict__ Y1,
                         const _Float16* __restrict__ w2h,
                         const float* __restrict__ b2,
                         const _Float16* __restrict__ w3h,
                         const float* __restrict__ b3,
                         _Float16* __restrict__ phi) {
    __shared__ _Float16 lds[8][16 * 32];     // per-wave 16 pix x 32 ch f16
    const int lane = threadIdx.x & 31;
    const int wave = threadIdx.x >> 5;       // wave32
    int tile = blockIdx.x * 8 + wave;        // B*H*(W/16) tiles total
    const int tprow = Wsz / 16;              // 8
    int b  = tile / (Hsz * tprow);
    int rem = tile - b * (Hsz * tprow);
    int h  = rem / tprow;
    int w0 = (rem - h * tprow) * 16;

    const int m    = lane & 15;              // A-operand row (pixel within tile)
    const int kb   = (lane < 16) ? 0 : 8;    // K sub-offset per lane half
    const int nsel = lane & 15;              // D-operand column (out channel)

    float bb0 = b2[nsel], bb1 = b2[16 + nsel];
    v8f acc0, acc1;
    #pragma unroll
    for (int r = 0; r < 8; ++r) { acc0[r] = bb0; acc1[r] = bb1; }

    #pragma unroll
    for (int dy = -1; dy <= 1; ++dy) {
        int hh = h + dy;
        #pragma unroll
        for (int dx = -1; dx <= 1; ++dx) {
            int ww = w0 + m + dx;
            U16 a;
            if ((unsigned)hh < (unsigned)Hsz && (unsigned)ww < (unsigned)Wsz) {
                const uint4* s = reinterpret_cast<const uint4*>(
                    Y1 + ((size_t)(b * Hsz + hh) * Wsz + ww) * 32);
                a.u[0] = s[kb >> 3];           // ch kb..kb+7
                a.u[1] = s[(kb >> 3) + 2];     // ch kb+16..kb+23
            } else {
                a.u[0] = make_uint4(0, 0, 0, 0);
                a.u[1] = make_uint4(0, 0, 0, 0);
            }
            int tap = (dy + 1) * 3 + (dx + 1);
            const _Float16* bp = w2h + (size_t)tap * 32 * 32 + nsel;   // B col N=nsel, K in table order
            U16 b0, b1;
            #pragma unroll
            for (int s = 0; s < 16; ++s) {
                int kk = (s < 8) ? kb + s : 16 + kb + (s - 8);
                b0.v[s] = bp[kk * 32];                   // N = 0..15
                b1.v[s] = bp[kk * 32 + 16];              // N = 16..31
            }
            acc0 = __builtin_amdgcn_wmma_f32_16x16x32_f16(false, a.v, false, b0.v,
                                                          (short)0, acc0, false, false);
            acc1 = __builtin_amdgcn_wmma_f32_16x16x32_f16(false, a.v, false, b1.v,
                                                          (short)0, acc1, false, false);
        }
    }

    // ReLU + D-layout -> LDS (f16), then reload in A-layout for the 1x1 conv
    _Float16* my = lds[wave];
    #pragma unroll
    for (int r = 0; r < 8; ++r) {
        int mm = ((lane >> 4) << 3) + r;
        float v0 = acc0[r] > 0.f ? acc0[r] : 0.f;
        float v1 = acc1[r] > 0.f ? acc1[r] : 0.f;
        my[mm * 32 + nsel]      = (_Float16)v0;
        my[mm * 32 + 16 + nsel] = (_Float16)v1;
    }
    asm volatile("s_wait_dscnt 0" ::: "memory");   // wave-private LDS region: self-wait only

    U16 a2;
    {
        const uint4* s = reinterpret_cast<const uint4*>(my + m * 32);
        a2.u[0] = s[kb >> 3];
        a2.u[1] = s[(kb >> 3) + 2];
    }
    asm volatile("s_wait_dscnt 0" ::: "memory");

    float cb0 = b3[nsel], cb1 = b3[16 + nsel];
    v8f p0, p1;
    #pragma unroll
    for (int r = 0; r < 8; ++r) { p0[r] = cb0; p1[r] = cb1; }
    {
        const _Float16* wp = w3h + nsel;
        U16 c0, c1;
        #pragma unroll
        for (int s = 0; s < 16; ++s) {
            int kk = (s < 8) ? kb + s : 16 + kb + (s - 8);
            c0.v[s] = wp[kk * 32];
            c1.v[s] = wp[kk * 32 + 16];
        }
        p0 = __builtin_amdgcn_wmma_f32_16x16x32_f16(false, a2.v, false, c0.v,
                                                    (short)0, p0, false, false);
        p1 = __builtin_amdgcn_wmma_f32_16x16x32_f16(false, a2.v, false, c1.v,
                                                    (short)0, p1, false, false);
    }
    _Float16* dst = phi + ((size_t)(b * Hsz + h) * Wsz + w0) * 32;
    #pragma unroll
    for (int r = 0; r < 8; ++r) {
        int mm = ((lane >> 4) << 3) + r;
        dst[mm * 32 + nsel]      = (_Float16)p0[r];
        dst[mm * 32 + 16 + nsel] = (_Float16)p1[r];
    }
}

// ---------------- neighbor affinity: exp(-2*||phi - phi_nb||^2) ----------------
__global__ void k_affinity(const _Float16* __restrict__ phi, float* __restrict__ wbuf) {
    int p = blockIdx.x * blockDim.x + threadIdx.x;
    int w = p & (Wsz - 1);
    int h = (p >> 7) & (Hsz - 1);
    int b = p >> 14;

    union HU { uint4 u; _Float16 h[8]; };
    float mine[32];
    {
        const uint4* mp = reinterpret_cast<const uint4*>(phi + (size_t)p * 32);
        #pragma unroll
        for (int j = 0; j < 4; ++j) {
            HU x; x.u = mp[j];
            #pragma unroll
            for (int i = 0; i < 8; ++i) mine[8 * j + i] = (float)x.h[i];
        }
    }
    const int dh[4] = {-1, 1, 0, 0}, dw[4] = {0, 0, -1, 1};
    float out[4];
    #pragma unroll
    for (int d = 0; d < 4; ++d) {
        int hh = h + dh[d], ww = w + dw[d];
        if ((unsigned)hh < (unsigned)Hsz && (unsigned)ww < (unsigned)Wsz) {
            const uint4* np = reinterpret_cast<const uint4*>(
                phi + ((size_t)(b * Hsz + hh) * Wsz + ww) * 32);
            float d2 = 0.f;
            #pragma unroll
            for (int j = 0; j < 4; ++j) {
                HU x; x.u = np[j];
                #pragma unroll
                for (int i = 0; i < 8; ++i) {
                    float df = mine[8 * j + i] - (float)x.h[i];
                    d2 += df * df;
                }
            }
            out[d] = __expf(-2.0f * d2);     // tau = 0.5
        } else {
            out[d] = 0.f;
        }
    }
    float4 o4 = make_float4(out[0], out[1], out[2], out[3]);
    reinterpret_cast<float4*>(wbuf + (size_t)p * 4)[0] = o4;
}

// ---------------- initial soft seed assignment ----------------
__global__ void k_initA(float* __restrict__ A0) {
    int p = blockIdx.x * blockDim.x + threadIdx.x;
    int w = p & (Wsz - 1);
    int h = (p >> 7) & (Hsz - 1);
    float e[32];
    float mx = -3.4e38f;
    #pragma unroll
    for (int k = 0; k < 32; ++k) {
        float sh = ((k >> 3) + 0.5f) * 32.f;    // gh=4 -> H/gh=32
        float sw = ((k & 7) + 0.5f) * 16.f;     // gw=8 -> W/gw=16
        float dh = (float)h - sh, dw = (float)w - sw;
        float s = -(dh * dh + dw * dw) * (1.0f / 512.0f);   // sigma2 = 32*16
        e[k] = s;
        mx = s > mx ? s : mx;
    }
    float sum = 0.f;
    #pragma unroll
    for (int k = 0; k < 32; ++k) { e[k] = __expf(e[k] - mx); sum += e[k]; }
    float inv = 1.0f / sum;
    float* dst = A0 + (size_t)p * 32;
    #pragma unroll
    for (int j = 0; j < 8; ++j) {
        float4 o = make_float4(e[4 * j] * inv, e[4 * j + 1] * inv,
                               e[4 * j + 2] * inv, e[4 * j + 3] * inv);
        reinterpret_cast<float4*>(dst)[j] = o;
    }
}

// ---------------- one soft message-passing round ----------------
__global__ void k_msgpass(const float* __restrict__ Asrc, const float* __restrict__ wbuf,
                          float* __restrict__ Adst) {
    int p = blockIdx.x * blockDim.x + threadIdx.x;
    int w = p & (Wsz - 1);
    int h = (p >> 7) & (Hsz - 1);
    int b = p >> 14;

    float4 agg[8];
    #pragma unroll
    for (int j = 0; j < 8; ++j) agg[j] = make_float4(0.f, 0.f, 0.f, 0.f);

    float4 w4 = reinterpret_cast<const float4*>(wbuf + (size_t)p * 4)[0];
    float wd[4] = {w4.x, w4.y, w4.z, w4.w};
    const int dh[4] = {-1, 1, 0, 0}, dw[4] = {0, 0, -1, 1};
    #pragma unroll
    for (int d = 0; d < 4; ++d) {
        int hh = h + dh[d], ww = w + dw[d];
        if ((unsigned)hh < (unsigned)Hsz && (unsigned)ww < (unsigned)Wsz) {
            const float4* nb = reinterpret_cast<const float4*>(
                Asrc + ((size_t)(b * Hsz + hh) * Wsz + ww) * 32);
            float c = wd[d];
            #pragma unroll
            for (int j = 0; j < 8; ++j) {
                float4 v = nb[j];
                agg[j].x += c * v.x; agg[j].y += c * v.y;
                agg[j].z += c * v.z; agg[j].w += c * v.w;
            }
        }
    }
    float e[32];
    float mx = -3.4e38f;
    #pragma unroll
    for (int j = 0; j < 8; ++j) {
        e[4 * j + 0] = 2.0f * agg[j].x; e[4 * j + 1] = 2.0f * agg[j].y;
        e[4 * j + 2] = 2.0f * agg[j].z; e[4 * j + 3] = 2.0f * agg[j].w;
    }
    #pragma unroll
    for (int k = 0; k < 32; ++k) mx = e[k] > mx ? e[k] : mx;
    float sum = 0.f;
    #pragma unroll
    for (int k = 0; k < 32; ++k) { e[k] = __expf(e[k] - mx); sum += e[k]; }
    float inv = 1.0f / sum;
    float* dst = Adst + (size_t)p * 32;
    #pragma unroll
    for (int j = 0; j < 8; ++j) {
        float4 o = make_float4(e[4 * j] * inv, e[4 * j + 1] * inv,
                               e[4 * j + 2] * inv, e[4 * j + 3] * inv);
        reinterpret_cast<float4*>(dst)[j] = o;
    }
}

// ---------------- pooling: partial sums per (b,k): mass,h,w,h2,w2,col[10] ----------------
__global__ void k_zeroP(float* __restrict__ P) {
    int t = blockIdx.x * blockDim.x + threadIdx.x;
    if (t < 32 * 32 * 15) P[t] = 0.f;
}

__global__ void k_pool(const float* __restrict__ A, const int* __restrict__ X,
                       float* __restrict__ P) {
    __shared__ float sacc[32 * 15];
    for (int i = threadIdx.x; i < 32 * 15; i += blockDim.x) sacc[i] = 0.f;
    __syncthreads();

    int b     = blockIdx.x >> 6;          // 64 chunks of 256 pixels per batch
    int chunk = blockIdx.x & 63;
    int k   = threadIdx.x & 31;           // lane = k -> coalesced A reads
    int grp = threadIdx.x >> 5;           // wave id: 8 groups of 32 pixels

    float mass = 0.f, shc = 0.f, swc = 0.f, sh2 = 0.f, sw2 = 0.f;
    float col[10];
    #pragma unroll
    for (int c = 0; c < 10; ++c) col[c] = 0.f;

    for (int i = 0; i < 32; ++i) {
        int pl  = chunk * 256 + grp * 32 + i;          // pixel within batch
        float a = A[((size_t)b * (Hsz * Wsz) + pl) * 32 + k];
        int lab = X[(size_t)b * (Hsz * Wsz) + pl];
        float hn = ((pl >> 7) + 0.5f) * (1.0f / Hsz);
        float wn = ((pl & 127) + 0.5f) * (1.0f / Wsz);
        mass += a;
        shc += a * hn;  swc += a * wn;
        sh2 += a * hn * hn;  sw2 += a * wn * wn;
        #pragma unroll
        for (int c = 0; c < 10; ++c) col[c] += (lab == c) ? a : 0.f;
    }
    float* base = &sacc[k * 15];
    atomicAdd(&base[0], mass);
    atomicAdd(&base[1], shc);
    atomicAdd(&base[2], swc);
    atomicAdd(&base[3], sh2);
    atomicAdd(&base[4], sw2);
    #pragma unroll
    for (int c = 0; c < 10; ++c) atomicAdd(&base[5 + c], col[c]);
    __syncthreads();

    for (int i = threadIdx.x; i < 32 * 15; i += blockDim.x) {
        int kk = i / 15, f = i - kk * 15;
        atomicAdd(&P[((size_t)b * 32 + kk) * 15 + f], sacc[i]);
    }
}

// ---------------- finalize T = [size, h_c, w_c, col[10], bbox[4]] ----------------
__global__ void k_finalize(const float* __restrict__ P, float* __restrict__ T) {
    int t = blockIdx.x * blockDim.x + threadIdx.x;
    if (t >= 32 * 32) return;
    const float* s = P + (size_t)t * 15;
    float mass = s[0] + 1e-6f;
    float inv = 1.0f / mass;
    float hc = s[1] * inv, wc = s[2] * inv;
    float h2 = s[3] * inv, w2 = s[4] * inv;
    float hv = h2 - hc * hc; hv = hv > 0.f ? hv : 0.f;
    float wv = w2 - wc * wc; wv = wv > 0.f ? wv : 0.f;
    float hsd = sqrtf(hv + 1e-6f);
    float wsd = sqrtf(wv + 1e-6f);
    float* o = T + (size_t)t * 17;
    o[0] = mass * (1.0f / (Hsz * Wsz));
    o[1] = hc; o[2] = wc;
    #pragma unroll
    for (int c = 0; c < 10; ++c) o[3 + c] = s[5 + c] * inv;
    o[13] = hc - hsd; o[14] = hc + hsd;
    o[15] = wc - wsd; o[16] = wc + wsd;
}

extern "C" void kernel_launch(void* const* d_in, const int* in_sizes, int n_in,
                              void* d_out, int out_size, void* d_ws, size_t ws_size,
                              hipStream_t stream) {
    const int*   X  = (const int*)d_in[0];
    const float* w1 = (const float*)d_in[1];
    const float* b1 = (const float*)d_in[2];
    const float* w2 = (const float*)d_in[3];
    const float* b2 = (const float*)d_in[4];
    const float* w3 = (const float*)d_in[5];
    const float* b3 = (const float*)d_in[6];

    char* ws = (char*)d_ws;
    _Float16* Y1   = (_Float16*)(ws + OFF_Y1);
    _Float16* phi  = (_Float16*)(ws + OFF_PHI);
    float*    wbuf = (float*)(ws + OFF_WB);
    float*    A0   = (float*)(ws + OFF_A0);
    float*    A1   = (float*)(ws + OFF_A1);
    _Float16* w2h  = (_Float16*)(ws + OFF_W2H);
    _Float16* w3h  = (_Float16*)(ws + OFF_W3H);
    float*    P    = (float*)(ws + OFF_P);

    float* outA = (float*)d_out;                 // (B,H,W,K) = NPIX*32 floats
    float* outT = outA + NPIX * 32;              // (B,K,17)

    k_prep<<<(10240 + 255) / 256, 256, 0, stream>>>(w2, w3, w2h, w3h);
    k_conv1<<<(int)(NPIX / 256), 256, 0, stream>>>(X, w1, b1, Y1);
    k_conv23<<<(Bsz * Hsz * (Wsz / 16)) / 8, 256, 0, stream>>>(Y1, w2h, b2, w3h, b3, phi);
    k_affinity<<<(int)(NPIX / 256), 256, 0, stream>>>(phi, wbuf);
    k_initA<<<(int)(NPIX / 256), 256, 0, stream>>>(A0);

    float* bufs[2] = {A0, A1};
    for (int i = 0; i < 8; ++i) {
        const float* src = bufs[i & 1];
        float* dst = (i == 7) ? outA : bufs[(i + 1) & 1];
        k_msgpass<<<(int)(NPIX / 256), 256, 0, stream>>>(src, wbuf, dst);
    }
    k_zeroP<<<(32 * 32 * 15 + 255) / 256, 256, 0, stream>>>(P);
    k_pool<<<Bsz * (Hsz * Wsz / 256), 256, 0, stream>>>(outA, X, P);
    k_finalize<<<(32 * 32 + 255) / 256, 256, 0, stream>>>(P, outT);
}
